// XorNeuronMLP_v2_32315333935410
// MI455X (gfx1250) — matrix operand order repaired and checked by reading, not compiled, not run on hardware
//
#include <hip/hip_runtime.h>
#include <hip/hip_bf16.h>

typedef __attribute__((ext_vector_type(16))) __bf16 v16bf;
typedef __attribute__((ext_vector_type(8)))  __bf16 v8bf;
typedef __attribute__((ext_vector_type(8)))  float  v8f;

#define BATCH   4096
#define IN_DIM  1024
#define HID     4096
#define PREV    512
#define T_CT    16
#define CG      32
#define IH      64
#define NCLS    10

// ---------------------------------------------------------------------------
// f32 -> bf16 conversion
// ---------------------------------------------------------------------------
__global__ __launch_bounds__(256)
void cvt_kernel(const float* __restrict__ s, __bf16* __restrict__ d, int n) {
    int i = blockIdx.x * 256 + threadIdx.x;
    if (i < n) d[i] = (__bf16)s[i];
}

// ---------------------------------------------------------------------------
// Tiled bf16 WMMA GEMM: P[M,N] = A[M,K] @ B[K,N] + bias[N]  (f32 out)
// Block tile 128x128, BK=32, 8 waves of 32x64 each (2x4 wmma accumulators).
// Double-buffered LDS staging, pre-swizzled into wave32 WMMA fragment order
// so each fragment load is 2x ds_load_b128. One barrier per k-step; next-tile
// global loads are issued before the WMMAs and committed to the alternate LDS
// buffer afterwards, hiding HBM/L2 latency behind the matrix pipe.
// ---------------------------------------------------------------------------
__global__ __launch_bounds__(256)
void gemm_bias_kernel(const __bf16* __restrict__ A, const __bf16* __restrict__ Bm,
                      const float* __restrict__ bias, float* __restrict__ P,
                      int M, int N, int K) {
    __shared__ __bf16 Abuf[2 * 8 * 512];  // double-buffered A frags (16 KB)
    __shared__ __bf16 Bbuf[2 * 8 * 512];  // double-buffered B frags (16 KB)

    const int tid  = threadIdx.x;
    const int lane = tid & 31;
    const int wave = tid >> 5;
    const int bm = blockIdx.y * 128;
    const int bn = blockIdx.x * 128;
    const int wm = wave >> 1;   // 0..3  (M direction, 2 subtiles each)
    const int wn = wave & 1;    // 0..1  (N direction, 4 subtiles each)

    v8f acc[2][4];
#pragma unroll
    for (int j = 0; j < 2; ++j)
#pragma unroll
        for (int q = 0; q < 4; ++q)
#pragma unroll
            for (int v = 0; v < 8; ++v) acc[j][q][v] = 0.0f;

    // staging coordinates
    const int asub  = tid >> 5;                    // A subtile 0..7
    const int am    = bm + asub * 16 + (lane & 15);
    const int akoff = (lane >> 4) * 16;
    const int bk    = tid >> 3;                    // B tile row 0..31
    const int bsub  = tid & 7;                     // B N-subtile 0..7
    const int bn0   = bn + bsub * 16;
    const int bi    = bk & 15;                     // half index within frag
    const int bLhi  = (bk >> 4) * 16;              // lane-group base

    v8bf ra0, ra1, rb0, rb1;                       // in-flight staging regs

    auto gload = [&](int kb) {
        const __bf16* sa = A + (size_t)am * K + kb + akoff;
        ra0 = *(const v8bf*)(sa);
        ra1 = *(const v8bf*)(sa + 8);
        const __bf16* sb = Bm + (size_t)(kb + bk) * N + bn0;
        rb0 = *(const v8bf*)(sb);
        rb1 = *(const v8bf*)(sb + 8);
        // prefetch two tiles ahead into the cache hierarchy
        if (kb + 32 < K) {
            __builtin_prefetch(A + (size_t)am * K + kb + 32 + akoff, 0, 1);
            __builtin_prefetch(Bm + (size_t)(kb + 32 + bk) * N + bn0, 0, 1);
        }
    };
    auto lstore = [&](int buf) {
        __bf16* d = &Abuf[buf * 4096 + asub * 512 + lane * 16];
        *(v8bf*)(d)     = ra0;
        *(v8bf*)(d + 8) = ra1;
        __bf16* base = &Bbuf[buf * 4096 + bsub * 512 + bi];
#pragma unroll
        for (int j = 0; j < 8; ++j) base[(bLhi + j) * 16]     = rb0[j];
#pragma unroll
        for (int j = 0; j < 8; ++j) base[(bLhi + 8 + j) * 16] = rb1[j];
    };

    gload(0);
    lstore(0);
    __syncthreads();

    int cur = 0;
    for (int kb = 0; kb < K; kb += 32) {
        const bool more = (kb + 32) < K;
        if (more) gload(kb + 32);                  // overlap with this tile's math

        v16bf af[2], bf_[4];
#pragma unroll
        for (int j = 0; j < 2; ++j)
            af[j] = *(const v16bf*)&Abuf[cur * 4096 + (wm * 2 + j) * 512 + lane * 16];
#pragma unroll
        for (int q = 0; q < 4; ++q)
            bf_[q] = *(const v16bf*)&Bbuf[cur * 4096 + (wn * 4 + q) * 512 + lane * 16];

#pragma unroll
        for (int j = 0; j < 2; ++j)
#pragma unroll
            for (int q = 0; q < 4; ++q)
                acc[j][q] = __builtin_amdgcn_wmma_f32_16x16x32_bf16(
                    false, af[j], false, bf_[q], (short)0, acc[j][q], false, false);

        if (more) lstore(cur ^ 1);                 // commit next tile while WMMAs run
        __syncthreads();
        cur ^= 1;
    }

    // epilogue: fused bias, f32 store
#pragma unroll
    for (int j = 0; j < 2; ++j) {
        const int mrow = bm + (wm * 2 + j) * 16 + (lane >> 4) * 8;
#pragma unroll
        for (int q = 0; q < 4; ++q) {
            const int n = bn + (wn * 4 + q) * 16 + (lane & 15);
            const float bv = bias[n];
#pragma unroll
            for (int v = 0; v < 8; ++v)
                P[(size_t)(mrow + v) * N + n] = acc[j][q][v] + bv;
        }
    }
}

// ---------------------------------------------------------------------------
// LayerNorm over rows of P (f32), write bf16
// ---------------------------------------------------------------------------
__global__ __launch_bounds__(256)
void ln_kernel(const float* __restrict__ P, __bf16* __restrict__ Z, int N) {
    __shared__ float red[16];
    const int row = blockIdx.x;
    const float* p = P + (size_t)row * N;
    float s = 0.0f, ss = 0.0f;
    for (int i = threadIdx.x; i < N; i += 256) {
        float x = p[i];
        s += x; ss += x * x;
    }
#pragma unroll
    for (int off = 16; off > 0; off >>= 1) {
        s  += __shfl_xor(s,  off, 32);
        ss += __shfl_xor(ss, off, 32);
    }
    const int wave = threadIdx.x >> 5, lane = threadIdx.x & 31;
    if (lane == 0) { red[wave * 2] = s; red[wave * 2 + 1] = ss; }
    __syncthreads();
    s = 0.0f; ss = 0.0f;
#pragma unroll
    for (int w = 0; w < 8; ++w) { s += red[w * 2]; ss += red[w * 2 + 1]; }
    const float mu  = s / (float)N;
    const float var = ss / (float)N - mu * mu;
    const float r   = rsqrtf(var + 1e-5f);
    __bf16* z = Z + (size_t)row * N;
    for (int i = threadIdx.x; i < N; i += 256) z[i] = (__bf16)((p[i] - mu) * r);
}

// ---------------------------------------------------------------------------
// Inner per-cell-type MLPs: z(8) -> relu 64 -> relu 64 -> 1, all via WMMA for
// layers 1/2 (K padded 8->32 for layer 1), shfl reduction for layer 3.
// Grid: 16 cell types x 1024 blocks; each of 8 waves does a 16-cell strip.
// ---------------------------------------------------------------------------
__global__ __launch_bounds__(256)
void inner_kernel(const __bf16* __restrict__ Z, const __bf16* __restrict__ W1b,
                  const float* __restrict__ b1, const __bf16* __restrict__ W2b,
                  const float* __restrict__ b2, const float* __restrict__ W3,
                  const float* __restrict__ b3, __bf16* __restrict__ Y) {
    __shared__ __bf16 W1lds[4 * 512];   // 4 N-subtile frags (K zero-padded)
    __shared__ __bf16 W2lds[8 * 512];   // (2 ksteps x 4 N-subtiles) frags
    __shared__ __bf16 H1[8 * 1024];     // per-wave 16x64 hidden activations

    const int tid  = threadIdx.x;
    const int lane = tid & 31;
    const int wave = tid >> 5;
    const int t    = blockIdx.x >> 10;      // cell type
    const int blk  = blockIdx.x & 1023;
    const int n_   = lane & 15;
    const int g    = lane >> 4;

    // ---- stage W2 fragments (64x64) into LDS ----
    {
        const int k = tid >> 2, sn = tid & 3;
        const __bf16* src = W2b + t * 4096 + k * 64 + sn * 16;
        v8bf r0 = *(const v8bf*)src;
        v8bf r1 = *(const v8bf*)(src + 8);
        const int f   = (k >> 5) * 4 + sn;
        const int i_  = k & 15;
        const int Lhi = ((k >> 4) & 1) * 16;
        __bf16* base = &W2lds[f * 512 + i_];
#pragma unroll
        for (int j = 0; j < 8; ++j) base[(Lhi + j) * 16]     = r0[j];
#pragma unroll
        for (int j = 0; j < 8; ++j) base[(Lhi + 8 + j) * 16] = r1[j];
    }
    // ---- zero W1 fragment region (K rows 8..31 are padding) ----
    {
        v8bf zz;
#pragma unroll
        for (int i = 0; i < 8; ++i) zz[i] = (__bf16)0.0f;
        *(v8bf*)&W1lds[tid * 8] = zz;
    }
    __syncthreads();
    // ---- fill valid W1 rows (8 x 64) ----
    if (tid < 128) {
        const int k = tid >> 4, j = tid & 15;
#pragma unroll
        for (int sn = 0; sn < 4; ++sn)
            W1lds[sn * 512 + j * 16 + k] = W1b[t * 512 + k * 64 + sn * 16 + j];
    }
    __syncthreads();

    // per-lane bias / W3 scalars for C-layout columns
    float b1v[4], b2v[4], w3v[4];
#pragma unroll
    for (int sn = 0; sn < 4; ++sn) {
        b1v[sn] = b1[t * 64 + sn * 16 + n_];
        b2v[sn] = b2[t * 64 + sn * 16 + n_];
        w3v[sn] = W3[t * 64 + sn * 16 + n_];
    }
    const float b3v = b3[t];

    v16bf w1f[4], w2f[8];
#pragma unroll
    for (int sn = 0; sn < 4; ++sn)
        w1f[sn] = *(const v16bf*)&W1lds[sn * 512 + lane * 16];
#pragma unroll
    for (int f = 0; f < 8; ++f)
        w2f[f] = *(const v16bf*)&W2lds[f * 512 + lane * 16];

    const int strip = blk * 8 + wave;   // 16-cell strip index within type t

    // ---- layer 1 A fragment (K=8 valid, rest zero) ----
    v16bf af;
#pragma unroll
    for (int i = 0; i < 16; ++i) af[i] = (__bf16)0.0f;
    if (g == 0) {
        const int r = strip * 16 + n_;
        const int b = r >> 5, c = r & 31;
        v8bf z8 = *(const v8bf*)(Z + (size_t)b * HID + t * 256 + c * 8);
#pragma unroll
        for (int i = 0; i < 8; ++i) af[i] = z8[i];
    }

    v8f acc[4];
#pragma unroll
    for (int sn = 0; sn < 4; ++sn) {
#pragma unroll
        for (int v = 0; v < 8; ++v) acc[sn][v] = 0.0f;
        acc[sn] = __builtin_amdgcn_wmma_f32_16x16x32_bf16(
            false, af, false, w1f[sn], (short)0, acc[sn], false, false);
    }

    // relu + bias, transpose through per-wave LDS into A-fragment layout
    __bf16* h1 = &H1[wave * 1024];
#pragma unroll
    for (int sn = 0; sn < 4; ++sn)
#pragma unroll
        for (int v = 0; v < 8; ++v) {
            float val = fmaxf(acc[sn][v] + b1v[sn], 0.0f);
            h1[(v + g * 8) * 64 + sn * 16 + n_] = (__bf16)val;
        }
    __syncthreads();

    v16bf hf[2];
#pragma unroll
    for (int kk = 0; kk < 2; ++kk) {
        const __bf16* base = &h1[n_ * 64 + kk * 32 + g * 16];
        v8bf r0 = *(const v8bf*)(base);
        v8bf r1 = *(const v8bf*)(base + 8);
#pragma unroll
        for (int i = 0; i < 8; ++i) { hf[kk][i] = r0[i]; hf[kk][8 + i] = r1[i]; }
    }

    // ---- layer 2: 16x64 @ 64x64, two K-steps ----
    v8f acc2[4];
#pragma unroll
    for (int sn = 0; sn < 4; ++sn) {
#pragma unroll
        for (int v = 0; v < 8; ++v) acc2[sn][v] = 0.0f;
#pragma unroll
        for (int kk = 0; kk < 2; ++kk)
            acc2[sn] = __builtin_amdgcn_wmma_f32_16x16x32_bf16(
                false, hf[kk], false, w2f[kk * 4 + sn], (short)0, acc2[sn], false, false);
    }

    // ---- layer 3: relu + dot with W3, cross-lane reduce over N ----
    float p[8];
#pragma unroll
    for (int v = 0; v < 8; ++v) {
        float s = 0.0f;
#pragma unroll
        for (int sn = 0; sn < 4; ++sn)
            s += fmaxf(acc2[sn][v] + b2v[sn], 0.0f) * w3v[sn];
        p[v] = s;
    }
#pragma unroll
    for (int off = 1; off < 16; off <<= 1)
#pragma unroll
        for (int v = 0; v < 8; ++v) p[v] += __shfl_xor(p[v], off, 32);

    if (n_ == 0) {
#pragma unroll
        for (int v = 0; v < 8; ++v) {
            const int r = strip * 16 + g * 8 + v;
            const int b = r >> 5, c = r & 31;
            Y[(size_t)b * PREV + t * 32 + c] = (__bf16)(p[v] + b3v);
        }
    }
}

// ---------------------------------------------------------------------------
// Head: logits = Y @ Wfc + bfc ; per-row NLL (log-softmax)
// ---------------------------------------------------------------------------
__global__ __launch_bounds__(256)
void head_kernel(const __bf16* __restrict__ Y, const float* __restrict__ Wfc,
                 const float* __restrict__ bfc, const int* __restrict__ labels,
                 float* __restrict__ logits, float* __restrict__ nll) {
    const int b = blockIdx.x * 256 + threadIdx.x;
    if (b >= BATCH) return;
    float acc[NCLS];
#pragma unroll
    for (int c = 0; c < NCLS; ++c) acc[c] = bfc[c];
    const __bf16* row = Y + (size_t)b * PREV;
    for (int k = 0; k < PREV; ++k) {
        const float a = (float)row[k];
        const float* w = Wfc + k * NCLS;
#pragma unroll
        for (int c = 0; c < NCLS; ++c) acc[c] += a * w[c];
    }
    float mx = acc[0];
#pragma unroll
    for (int c = 1; c < NCLS; ++c) mx = fmaxf(mx, acc[c]);
    float se = 0.0f;
#pragma unroll
    for (int c = 0; c < NCLS; ++c) se += __expf(acc[c] - mx);
    const float lse = __logf(se) + mx;
#pragma unroll
    for (int c = 0; c < NCLS; ++c) logits[(size_t)b * NCLS + c] = acc[c];
    const int lab = labels[b];
    nll[b] = lse - acc[lab];
}

__global__ __launch_bounds__(256)
void loss_kernel(const float* __restrict__ nll, float* __restrict__ out) {
    __shared__ float red[256];
    float s = 0.0f;
    for (int i = threadIdx.x; i < BATCH; i += 256) s += nll[i];
    red[threadIdx.x] = s;
    __syncthreads();
    for (int off = 128; off > 0; off >>= 1) {
        if (threadIdx.x < off) red[threadIdx.x] += red[threadIdx.x + off];
        __syncthreads();
    }
    if (threadIdx.x == 0) out[0] = red[0] / (float)BATCH;
}

// ---------------------------------------------------------------------------
extern "C" void kernel_launch(void* const* d_in, const int* in_sizes, int n_in,
                              void* d_out, int out_size, void* d_ws, size_t ws_size,
                              hipStream_t stream) {
    const float* x     = (const float*)d_in[0];
    const int*   labels= (const int*)  d_in[1];
    const float* Wout0 = (const float*)d_in[2];
    const float* bout0 = (const float*)d_in[3];
    const float* Wout1 = (const float*)d_in[4];
    const float* bout1 = (const float*)d_in[5];
    const float* Wout2 = (const float*)d_in[6];
    const float* bout2 = (const float*)d_in[7];
    const float* W1    = (const float*)d_in[8];
    const float* b1    = (const float*)d_in[9];
    const float* W2    = (const float*)d_in[10];
    const float* b2    = (const float*)d_in[11];
    const float* W3    = (const float*)d_in[12];
    const float* b3    = (const float*)d_in[13];
    const float* Wfc   = (const float*)d_in[14];
    const float* bfc   = (const float*)d_in[15];

    char* ws = (char*)d_ws;
    size_t off = 0;
    auto alloc = [&](size_t bytes) -> void* {
        void* p = ws + off;
        off += (bytes + 255) & ~(size_t)255;
        return p;
    };
    __bf16* Xb   = (__bf16*)alloc((size_t)BATCH * IN_DIM * 2);
    __bf16* Wo0b = (__bf16*)alloc((size_t)IN_DIM * HID * 2);
    __bf16* Wo1b = (__bf16*)alloc((size_t)PREV * HID * 2);
    __bf16* Wo2b = (__bf16*)alloc((size_t)PREV * HID * 2);
    __bf16* W1b  = (__bf16*)alloc((size_t)T_CT * 8 * 64 * 2);
    __bf16* W2b  = (__bf16*)alloc((size_t)T_CT * 64 * 64 * 2);
    float*  Pbuf = (float*) alloc((size_t)BATCH * HID * 4);
    __bf16* Zb   = (__bf16*)alloc((size_t)BATCH * HID * 2);
    __bf16* Yb   = (__bf16*)alloc((size_t)BATCH * PREV * 2);
    float*  nll  = (float*) alloc((size_t)BATCH * 4);

    auto cvt = [&](const float* s, __bf16* d, int n) {
        cvt_kernel<<<(n + 255) / 256, 256, 0, stream>>>(s, d, n);
    };
    cvt(x, Xb, BATCH * IN_DIM);
    cvt(Wout0, Wo0b, IN_DIM * HID);
    cvt(Wout1, Wo1b, PREV * HID);
    cvt(Wout2, Wo2b, PREV * HID);
    cvt(W1, W1b, T_CT * 8 * 64);
    cvt(W2, W2b, T_CT * 64 * 64);

    const __bf16* Wo[3] = {Wo0b, Wo1b, Wo2b};
    const float*  bo[3] = {bout0, bout1, bout2};
    const __bf16* Aptr = Xb;
    int K = IN_DIM;
    for (int layer = 0; layer < 3; ++layer) {
        dim3 grid(HID / 128, BATCH / 128);
        gemm_bias_kernel<<<grid, 256, 0, stream>>>(Aptr, Wo[layer], bo[layer],
                                                   Pbuf, BATCH, HID, K);
        ln_kernel<<<BATCH, 256, 0, stream>>>(Pbuf, Zb, HID);
        inner_kernel<<<T_CT * 1024, 256, 0, stream>>>(Zb, W1b, b1, W2b, b2,
                                                      W3, b3, Yb);
        Aptr = Yb;
        K = PREV;
    }
    float* logits = (float*)d_out;
    head_kernel<<<BATCH / 256, 256, 0, stream>>>(Yb, Wfc, bfc, labels, logits, nll);
    loss_kernel<<<1, 256, 0, stream>>>(nll, logits + (size_t)BATCH * NCLS);
}
